// Convblock_72919954751797
// MI455X (gfx1250) — matrix operand order repaired and checked by reading, not compiled, not run on hardware
//
#include <hip/hip_runtime.h>
#include <hip/hip_bf16.h>

#define BB 8
#define SS 4096
#define CC 512
#define KK 7
#define RR (BB*SS)          // 32768 rows total
#define EPSBN 1e-5f

#define TM 128              // M tile (rows)
#define TN 128              // N tile (cols)
#define TK 32               // K chunk (bf16 wmma k=32)
#define LPAD 8              // LDS row padding (bf16 elems) -> 80B stride

typedef __bf16 bf16_t;
typedef __bf16 v8bf  __attribute__((ext_vector_type(8)));
typedef __bf16 v16bf __attribute__((ext_vector_type(16)));
typedef float  v8f   __attribute__((ext_vector_type(8)));

static __device__ inline v16bf pack16(v8bf lo, v8bf hi) {
    return __builtin_shufflevector(lo, hi, 0,1,2,3,4,5,6,7,8,9,10,11,12,13,14,15);
}

static __device__ inline float gelu_tanh(float v) {
    const float k0 = 0.7978845608028654f;
    return 0.5f * v * (1.f + tanhf(k0 * (v + 0.044715f * v * v * v)));
}

// CDNA5 async memory->LDS copy (16B per lane), tracked with ASYNCcnt.
// VDST VGPR holds the per-lane LDS byte offset; the low 32 bits of a generic
// __shared__ pointer are exactly that (FLAT LDS aperture truncation).
static __device__ inline void async_copy16(void* lds, const void* gptr) {
    uint32_t ldsoff = (uint32_t)(uintptr_t)lds;
    asm volatile("global_load_async_to_lds_b128 %0, %1, off"
                 :: "v"(ldsoff), "v"(gptr) : "memory");
}
static __device__ inline void async_wait0() {
    asm volatile("s_wait_asynccnt 0x0" ::: "memory");
}

// ---------------------------------------------------------------------------
// K0: transpose weights (d,c,k)->wt[k][d][c] bf16, conv_kernel (c,o)->ckT[o][c]
//     bf16, and zero the stat accumulators.
// ---------------------------------------------------------------------------
__global__ void prep_kernel(const float* __restrict__ w,
                            const float* __restrict__ ck,
                            bf16_t* __restrict__ wt,
                            bf16_t* __restrict__ ckT,
                            float* __restrict__ stats) {
    int tid = blockIdx.x * blockDim.x + threadIdx.x;
    int nthreads = gridDim.x * blockDim.x;
    if (tid < 4096) stats[tid] = 0.f;
    const int total = CC * CC;
    for (int i = tid; i < total; i += nthreads) {
        int d = i / CC, c = i % CC;
        const float* src = w + (size_t)(d * CC + c) * KK;
        #pragma unroll
        for (int k = 0; k < KK; ++k)
            wt[(size_t)k * CC * CC + (size_t)d * CC + c] = (bf16_t)src[k];
    }
    for (int i = tid; i < total; i += nthreads) {
        int o = i / CC, c = i % CC;
        ckT[(size_t)o * CC + c] = (bf16_t)ck[(size_t)c * CC + o];
    }
}

// ---------------------------------------------------------------------------
// K1: fused dynamic conv.
//   For tile (b, s-block, d-block): for each tap k:
//     w_k = X(s,c) @ Wk(d,c)^T   via v_wmma_f32_16x16x32_bf16
//     acc_y += tanh(w_k) * x[b, s+k-3, d]
//   y_pre = x + acc_y ; accumulate per-channel sum/sumsq for BN1.
//   Weight tiles stream memory->LDS via global_load_async_to_lds_b128.
// ---------------------------------------------------------------------------
__global__ __launch_bounds__(256) void dynconv_kernel(
        const float* __restrict__ x, const bf16_t* __restrict__ wt,
        float* __restrict__ ypre,
        float* __restrict__ bn1_sum, float* __restrict__ bn1_sq) {
    __shared__ bf16_t Xs[2][TM][TK + LPAD];
    __shared__ bf16_t Ws[2][TN][TK + LPAD];
    __shared__ float s_sum[TN];
    __shared__ float s_sq[TN];

    const int t    = threadIdx.x;
    const int lane = t & 31;
    const int wid  = t >> 5;
    const int wm0  = (wid >> 1) * 32;   // wave M origin in tile
    const int wn0  = (wid & 1) * 64;    // wave N origin in tile
    const int mblk = blockIdx.x;        // 0..255
    const int b    = mblk >> 5;         // batch
    const int st   = (mblk & 31) * TM;  // s offset within batch
    const int d0   = blockIdx.y * TN;   // channel tile origin

    if (t < TN) { s_sum[t] = 0.f; s_sq[t] = 0.f; }

    const float* xb = x + (size_t)b * SS * CC;
    const int rlo    = lane & 15;       // row/col within 16x16 frag
    const int khA    = (lane >> 4) * 8; // A frag K-half
    const int khB    = (lane >> 4) * 8; // B frag K-half
    const int mhalf  = (lane >> 4) * 8; // C/D M offset

    float accy[2][4][8];
    #pragma unroll
    for (int i = 0; i < 2; ++i)
        #pragma unroll
        for (int j = 0; j < 4; ++j)
            #pragma unroll
            for (int e = 0; e < 8; ++e) accy[i][j][e] = 0.f;

    for (int k = 0; k < KK; ++k) {
        const bf16_t* wk = wt + (size_t)k * CC * CC + (size_t)d0 * CC;

        v8f accw[2][4];
        #pragma unroll
        for (int i = 0; i < 2; ++i)
            #pragma unroll
            for (int j = 0; j < 4; ++j) accw[i][j] = (v8f){0,0,0,0,0,0,0,0};

        auto load_tiles = [&](int c0, int buf) {
            // X chunk: 128x32 fp32 -> bf16 via VGPRs (conversion required)
            #pragma unroll
            for (int it = 0; it < 4; ++it) {
                int lin = t + it * 256;           // 0..1023 float4 slots
                int row = lin >> 3;
                int c4  = (lin & 7) << 2;
                float4 v = *(const float4*)(xb + (size_t)(st + row) * CC + c0 + c4);
                bf16_t* dst = &Xs[buf][row][c4];
                dst[0] = (bf16_t)v.x; dst[1] = (bf16_t)v.y;
                dst[2] = (bf16_t)v.z; dst[3] = (bf16_t)v.w;
            }
            // W chunk: 128x32 bf16 — async DMA straight into LDS
            #pragma unroll
            for (int it = 0; it < 2; ++it) {
                int lin = t + it * 256;           // 0..511 16B slots
                int row = lin >> 2;
                int c8  = (lin & 3) << 3;
                async_copy16(&Ws[buf][row][c8],
                             wk + (size_t)row * CC + c0 + c8);
            }
        };

        int buf = 0;
        load_tiles(0, 0);
        async_wait0();
        __syncthreads();
        for (int cchunk = 0; cchunk < CC / TK; ++cchunk) {
            if (cchunk + 1 < CC / TK) load_tiles((cchunk + 1) * TK, buf ^ 1);
            // hint the L2 about the next tap's weights
            if (k + 1 < KK)
                __builtin_prefetch(wk + (size_t)CC * CC + (size_t)(t * 2) * CC, 0, 1);

            v16bf afrag[2];
            #pragma unroll
            for (int mt = 0; mt < 2; ++mt) {
                const bf16_t* p = &Xs[buf][wm0 + mt * 16 + rlo][khA];
                afrag[mt] = pack16(*(const v8bf*)p, *(const v8bf*)(p + 16));
            }
            #pragma unroll
            for (int nt = 0; nt < 4; ++nt) {
                const bf16_t* p = &Ws[buf][wn0 + nt * 16 + rlo][khB];
                v16bf bfrag = pack16(*(const v8bf*)p, *(const v8bf*)(p + 16));
                #pragma unroll
                for (int mt = 0; mt < 2; ++mt) {
                    accw[mt][nt] = __builtin_amdgcn_wmma_f32_16x16x32_bf16(
                        false, afrag[mt], false, bfrag,
                        (short)0, accw[mt][nt], false, false);
                }
            }
            async_wait0();       // async LDS writes for buf^1 must land
            __syncthreads();     // before any wave reads them
            buf ^= 1;
        }

        // gate with tanh and accumulate shifted-x product (im2col fused)
        #pragma unroll
        for (int mt = 0; mt < 2; ++mt) {
            #pragma unroll
            for (int e = 0; e < 8; ++e) {
                int srow = st + wm0 + mt * 16 + mhalf + e;
                int sp   = srow + k - 3;
                bool inb = (sp >= 0) && (sp < SS);
                const float* xr = xb + (size_t)(inb ? sp : 0) * CC + d0;
                #pragma unroll
                for (int nt = 0; nt < 4; ++nt) {
                    float xv = inb ? xr[wn0 + nt * 16 + rlo] : 0.f;
                    accy[mt][nt][e] += tanhf(accw[mt][nt][e]) * xv;
                }
            }
        }
    }

    // epilogue: skip add, store y_pre, BN1 partial stats
    #pragma unroll
    for (int mt = 0; mt < 2; ++mt) {
        #pragma unroll
        for (int e = 0; e < 8; ++e) {
            int srow = st + wm0 + mt * 16 + mhalf + e;
            const float* xr = xb + (size_t)srow * CC + d0;
            float* yr = ypre + ((size_t)b * SS + srow) * CC + d0;
            #pragma unroll
            for (int nt = 0; nt < 4; ++nt) {
                int nc = wn0 + nt * 16 + rlo;
                float yv = xr[nc] + accy[mt][nt][e];
                yr[nc] = yv;
                atomicAdd(&s_sum[nc], yv);
                atomicAdd(&s_sq[nc], yv * yv);
            }
        }
    }
    __syncthreads();
    if (t < TN) {
        atomicAdd(&bn1_sum[d0 + t], s_sum[t]);
        atomicAdd(&bn1_sq[d0 + t],  s_sq[t]);
    }
}

// ---------------------------------------------------------------------------
// K2/K4: turn (sum, sumsq, scale, bias) into affine (a, b): xn = a*x + b
// ---------------------------------------------------------------------------
__global__ void finalize_stats(const float* __restrict__ sum,
                               const float* __restrict__ sq,
                               const float* __restrict__ scale,
                               const float* __restrict__ bias,
                               float* __restrict__ a, float* __restrict__ bo) {
    int c = blockIdx.x * blockDim.x + threadIdx.x;
    if (c >= CC) return;
    const float inv_n = 1.f / (float)RR;
    float mean = sum[c] * inv_n;
    float var  = sq[c] * inv_n - mean * mean;
    float g    = scale[c] * rsqrtf(var + EPSBN);
    a[c]  = g;
    bo[c] = bias[c] - mean * g;
}

// ---------------------------------------------------------------------------
// K3: z_pre = BN1(y_pre) @ conv_kernel ; BN1 folded into A-tile load;
//     accumulates BN2 channel stats. B tiles use async memory->LDS DMA.
// ---------------------------------------------------------------------------
__global__ __launch_bounds__(256) void conv1x1_kernel(
        const float* __restrict__ ypre, const bf16_t* __restrict__ ckT,
        const float* __restrict__ a1, const float* __restrict__ b1,
        float* __restrict__ zpre,
        float* __restrict__ bn2_sum, float* __restrict__ bn2_sq) {
    __shared__ bf16_t Ys[2][TM][TK + LPAD];
    __shared__ bf16_t Cs[2][TN][TK + LPAD];
    __shared__ float s_sum[TN];
    __shared__ float s_sq[TN];

    const int t    = threadIdx.x;
    const int lane = t & 31;
    const int wid  = t >> 5;
    const int wm0  = (wid >> 1) * 32;
    const int wn0  = (wid & 1) * 64;
    const int r0   = blockIdx.x * TM;     // 0..32640
    const int d0   = blockIdx.y * TN;

    if (t < TN) { s_sum[t] = 0.f; s_sq[t] = 0.f; }

    const int rlo   = lane & 15;
    const int khA   = (lane >> 4) * 8;
    const int khB   = (lane >> 4) * 8;
    const int mhalf = (lane >> 4) * 8;

    v8f accz[2][4];
    #pragma unroll
    for (int i = 0; i < 2; ++i)
        #pragma unroll
        for (int j = 0; j < 4; ++j) accz[i][j] = (v8f){0,0,0,0,0,0,0,0};

    auto load_tiles = [&](int c0, int buf) {
        #pragma unroll
        for (int it = 0; it < 4; ++it) {
            int lin = t + it * 256;
            int row = lin >> 3;
            int c4  = (lin & 7) << 2;
            float4 v  = *(const float4*)(ypre + (size_t)(r0 + row) * CC + c0 + c4);
            float4 ga = *(const float4*)(a1 + c0 + c4);
            float4 gb = *(const float4*)(b1 + c0 + c4);
            bf16_t* dst = &Ys[buf][row][c4];
            dst[0] = (bf16_t)(v.x * ga.x + gb.x);
            dst[1] = (bf16_t)(v.y * ga.y + gb.y);
            dst[2] = (bf16_t)(v.z * ga.z + gb.z);
            dst[3] = (bf16_t)(v.w * ga.w + gb.w);
        }
        #pragma unroll
        for (int it = 0; it < 2; ++it) {
            int lin = t + it * 256;
            int row = lin >> 2;
            int c8  = (lin & 3) << 3;
            async_copy16(&Cs[buf][row][c8],
                         ckT + (size_t)(d0 + row) * CC + c0 + c8);
        }
    };

    int buf = 0;
    load_tiles(0, 0);
    async_wait0();
    __syncthreads();
    for (int cchunk = 0; cchunk < CC / TK; ++cchunk) {
        if (cchunk + 1 < CC / TK) load_tiles((cchunk + 1) * TK, buf ^ 1);

        v16bf afrag[2];
        #pragma unroll
        for (int mt = 0; mt < 2; ++mt) {
            const bf16_t* p = &Ys[buf][wm0 + mt * 16 + rlo][khA];
            afrag[mt] = pack16(*(const v8bf*)p, *(const v8bf*)(p + 16));
        }
        #pragma unroll
        for (int nt = 0; nt < 4; ++nt) {
            const bf16_t* p = &Cs[buf][wn0 + nt * 16 + rlo][khB];
            v16bf bfrag = pack16(*(const v8bf*)p, *(const v8bf*)(p + 16));
            #pragma unroll
            for (int mt = 0; mt < 2; ++mt) {
                accz[mt][nt] = __builtin_amdgcn_wmma_f32_16x16x32_bf16(
                    false, afrag[mt], false, bfrag,
                    (short)0, accz[mt][nt], false, false);
            }
        }
        async_wait0();
        __syncthreads();
        buf ^= 1;
    }

    #pragma unroll
    for (int mt = 0; mt < 2; ++mt) {
        #pragma unroll
        for (int e = 0; e < 8; ++e) {
            int row = r0 + wm0 + mt * 16 + mhalf + e;
            float* zr = zpre + (size_t)row * CC + d0;
            #pragma unroll
            for (int nt = 0; nt < 4; ++nt) {
                int nc = wn0 + nt * 16 + rlo;
                float zv = accz[mt][nt][e];
                zr[nc] = zv;
                atomicAdd(&s_sum[nc], zv);
                atomicAdd(&s_sq[nc], zv * zv);
            }
        }
    }
    __syncthreads();
    if (t < TN) {
        atomicAdd(&bn2_sum[d0 + t], s_sum[t]);
        atomicAdd(&bn2_sq[d0 + t],  s_sq[t]);
    }
}

// ---------------------------------------------------------------------------
// K5: out = BN1(y_pre) + gelu(BN2(z_pre)), vectorized float4.
// ---------------------------------------------------------------------------
__global__ void epilogue_kernel(const float* __restrict__ ypre,
                                const float* __restrict__ zpre,
                                const float* __restrict__ a1, const float* __restrict__ b1,
                                const float* __restrict__ a2, const float* __restrict__ b2,
                                float* __restrict__ out) {
    const size_t total4 = (size_t)RR * CC / 4;
    size_t stride = (size_t)gridDim.x * blockDim.x;
    for (size_t i = (size_t)blockIdx.x * blockDim.x + threadIdx.x; i < total4; i += stride) {
        int c0 = (int)((i * 4) & (CC - 1));
        float4 y = ((const float4*)ypre)[i];
        float4 z = ((const float4*)zpre)[i];
        float4 ga1 = *(const float4*)(a1 + c0);
        float4 gb1 = *(const float4*)(b1 + c0);
        float4 ga2 = *(const float4*)(a2 + c0);
        float4 gb2 = *(const float4*)(b2 + c0);
        float4 o;
        o.x = (y.x * ga1.x + gb1.x) + gelu_tanh(z.x * ga2.x + gb2.x);
        o.y = (y.y * ga1.y + gb1.y) + gelu_tanh(z.y * ga2.y + gb2.y);
        o.z = (y.z * ga1.z + gb1.z) + gelu_tanh(z.z * ga2.z + gb2.z);
        o.w = (y.w * ga1.w + gb1.w) + gelu_tanh(z.w * ga2.w + gb2.w);
        ((float4*)out)[i] = o;
    }
}

// ---------------------------------------------------------------------------
extern "C" void kernel_launch(void* const* d_in, const int* in_sizes, int n_in,
                              void* d_out, int out_size, void* d_ws, size_t ws_size,
                              hipStream_t stream) {
    const float* x   = (const float*)d_in[0];
    const float* w   = (const float*)d_in[1];
    const float* s1  = (const float*)d_in[2];
    const float* bi1 = (const float*)d_in[3];
    const float* ck  = (const float*)d_in[4];
    const float* s2  = (const float*)d_in[5];
    const float* bi2 = (const float*)d_in[6];
    float* out = (float*)d_out;

    char* ws = (char*)d_ws;
    float* stats   = (float*)ws;                  // 4096 floats (16 KB)
    float* bn1_sum = stats;
    float* bn1_sq  = stats + 512;
    float* a1      = stats + 1024;
    float* b1v     = stats + 1536;
    float* bn2_sum = stats + 2048;
    float* bn2_sq  = stats + 2560;
    float* a2      = stats + 3072;
    float* b2v     = stats + 3584;
    bf16_t* wt  = (bf16_t*)(ws + (size_t)1 * (1u << 20));   // 3.5 MB
    bf16_t* ckT = (bf16_t*)(ws + (size_t)5 * (1u << 20));   // 0.5 MB
    float* ypre = (float*)(ws + (size_t)8  * (1u << 20));   // 64 MB
    float* zpre = (float*)(ws + (size_t)72 * (1u << 20));   // 64 MB

    prep_kernel<<<512, 256, 0, stream>>>(w, ck, wt, ckT, stats);
    dynconv_kernel<<<dim3(256, 4), 256, 0, stream>>>(x, wt, ypre, bn1_sum, bn1_sq);
    finalize_stats<<<2, 256, 0, stream>>>(bn1_sum, bn1_sq, s1, bi1, a1, b1v);
    conv1x1_kernel<<<dim3(256, 4), 256, 0, stream>>>(ypre, ckT, a1, b1v, zpre,
                                                     bn2_sum, bn2_sq);
    finalize_stats<<<2, 256, 0, stream>>>(bn2_sum, bn2_sq, s2, bi2, a2, b2v);
    epilogue_kernel<<<4096, 256, 0, stream>>>(ypre, zpre, a1, b1v, a2, b2v, out);
}
